// FpgnnModel_25546465476978
// MI455X (gfx1250) — matrix operand order repaired and checked by reading, not compiled, not run on hardware
//
#include <hip/hip_runtime.h>
#include <hip/hip_bf16.h>
#include <cstdint>

// ---------------------------------------------------------------------------
// FP-GNN forward for MI455X (gfx1250, wave32, WMMA).
// All GEMMs run on v_wmma_f32_16x16x32_bf16 (f32 accumulate).
// Wave tile 32x64 (8 accumulators), software-pipelined K loop, prefetch.
// B operands are pre-transposed to bf16 so fragments load contiguously.
// ---------------------------------------------------------------------------

#define DEVFN __device__ __forceinline__

typedef __attribute__((ext_vector_type(16))) __bf16 v16bf;
typedef __attribute__((ext_vector_type(8)))  float  v8f;

static constexpr int Bsz   = 256;
static constexpr int Natom = 128;
static constexpr int Fdim  = 133;
static constexpr int Hh    = 8;
static constexpr int HID   = 300;
static constexpr int FP_DIM = 1489;

static constexpr int FPAD  = 160;   // Fdim padded to mult of 32
static constexpr int HNH   = 512;   // H * NHID
static constexpr int HIDP  = 320;   // HID padded to mult of 64
static constexpr int FPPAD = 1504;  // FP_DIM padded to mult of 32
static constexpr int CATP  = 608;   // 600 padded to mult of 32

static constexpr float NEGV = -9.0e15f;

DEVFN unsigned short f2bf(float x) {
  unsigned u = __float_as_uint(x);
  u += 0x7FFFu + ((u >> 16) & 1u);
  return (unsigned short)(u >> 16);
}

union FragU { v16bf v; unsigned short u[16]; };

// A fragment: 16x32 bf16, row-major source (lda in elements).
// ISA layout: lanes 0-15 row m=lane: VGPR0..3 = K 0..7, VGPR4..7 = K 16..23
//             lanes 16-31 row m=lane-16: K 8..15 and K 24..31.
DEVFN v16bf load_fragA(const unsigned short* p, int ld, int lane) {
  FragU f;
  int r  = lane & 15;
  int hi = (lane >> 4) << 3;          // 0 or 8
  const unsigned short* q = p + (size_t)r * ld;
#pragma unroll
  for (int i = 0; i < 8; ++i) {
    int kb = ((i & 4) << 2) + hi + ((i & 3) << 1);  // (i>=4?16:0)+hi+(i%4)*2
    f.u[2 * i]     = q[kb];
    f.u[2 * i + 1] = q[kb + 1];
  }
  return f.v;
}

// B fragment: 32x16 bf16, supplied TRANSPOSED (BT is N x K row-major, ldb = K
// stride). B layout matches A: lanes 0-15 col n=lane hold K 0..7 and 16..23,
// lanes 16-31 hold K 8..15 and 24..31.
DEVFN v16bf load_fragB(const unsigned short* bt, int ld, int lane) {
  FragU f;
  int n  = lane & 15;
  int kh = (lane >> 4) << 3;          // 0 or 8
  const unsigned short* q = bt + (size_t)n * ld + kh;
#pragma unroll
  for (int i = 0; i < 16; ++i) f.u[i] = q[((i & 8) << 1) + (i & 7)];
  return f.v;
}

#define WMMA_BF16(acc, a, b) \
  acc = __builtin_amdgcn_wmma_f32_16x16x32_bf16(false, a, false, b, (short)0, acc, false, false)

// Batched GEMM: C[z] = A[z] (MxK bf16) * B[z] (KxN via BT: NxK bf16), f32 out.
// z = bb*hdiv + hh; per-operand strides given separately for bb and hh.
// grid = (N/64, M/128, batch), block = 128 (4 waves; wave tile = 32x64).
__global__ void wmma_gemm(const unsigned short* __restrict__ A, int lda,
                          long long aB, long long aH,
                          const unsigned short* __restrict__ BT, int ldb,
                          long long bB, long long bH,
                          float* __restrict__ C, int ldc,
                          long long cB, long long cH,
                          int K, int hdiv) {
  int lane = threadIdx.x & 31;
  int wave = threadIdx.x >> 5;
  int z  = blockIdx.z;
  int bb = z / hdiv, hh = z % hdiv;
  size_t mrow = (size_t)blockIdx.y * 128 + (size_t)wave * 32;
  const unsigned short* Ab = A  + bb * aB + hh * aH + mrow * lda;
  const unsigned short* Bb = BT + bb * bB + hh * bH + (size_t)blockIdx.x * 64 * ldb;
  float* Cb = C + bb * cB + hh * cH + mrow * ldc + (size_t)blockIdx.x * 64;

  const unsigned short* Ab1 = Ab + (size_t)16 * lda;
  const unsigned short* Bb1 = Bb + (size_t)16 * ldb;
  const unsigned short* Bb2 = Bb + (size_t)32 * ldb;
  const unsigned short* Bb3 = Bb + (size_t)48 * ldb;

  v8f c00 = {}, c01 = {}, c02 = {}, c03 = {};
  v8f c10 = {}, c11 = {}, c12 = {}, c13 = {};

  // Preload k = 0 fragments.
  v16bf a0 = load_fragA(Ab,  lda, lane);
  v16bf a1 = load_fragA(Ab1, lda, lane);
  v16bf b0 = load_fragB(Bb,  ldb, lane);
  v16bf b1 = load_fragB(Bb1, ldb, lane);
  v16bf b2 = load_fragB(Bb2, ldb, lane);
  v16bf b3 = load_fragB(Bb3, ldb, lane);

  for (int k0 = 32; k0 < K; k0 += 32) {
    // Prefetch one K-block ahead of the in-flight loads (global_prefetch_b8).
    __builtin_prefetch(Ab + k0 + 32, 0, 1);
    __builtin_prefetch(Bb + k0 + 32, 0, 1);
    __builtin_prefetch(Bb2 + k0 + 32, 0, 1);
    // Issue next-K loads, then overlap them with 8 WMMAs on current frags.
    v16bf na0 = load_fragA(Ab  + k0, lda, lane);
    v16bf na1 = load_fragA(Ab1 + k0, lda, lane);
    v16bf nb0 = load_fragB(Bb  + k0, ldb, lane);
    v16bf nb1 = load_fragB(Bb1 + k0, ldb, lane);
    v16bf nb2 = load_fragB(Bb2 + k0, ldb, lane);
    v16bf nb3 = load_fragB(Bb3 + k0, ldb, lane);

    WMMA_BF16(c00, a0, b0);
    WMMA_BF16(c01, a0, b1);
    WMMA_BF16(c02, a0, b2);
    WMMA_BF16(c03, a0, b3);
    WMMA_BF16(c10, a1, b0);
    WMMA_BF16(c11, a1, b1);
    WMMA_BF16(c12, a1, b2);
    WMMA_BF16(c13, a1, b3);

    a0 = na0; a1 = na1;
    b0 = nb0; b1 = nb1; b2 = nb2; b3 = nb3;
  }
  // Epilogue on last fragments.
  WMMA_BF16(c00, a0, b0);
  WMMA_BF16(c01, a0, b1);
  WMMA_BF16(c02, a0, b2);
  WMMA_BF16(c03, a0, b3);
  WMMA_BF16(c10, a1, b0);
  WMMA_BF16(c11, a1, b1);
  WMMA_BF16(c12, a1, b2);
  WMMA_BF16(c13, a1, b3);

  // D layout: VGPR v -> rows v (lanes 0-15) and v+8 (lanes 16-31), col = lane&15.
  int n  = lane & 15;
  int mo = (lane >> 4) << 3;
#pragma unroll
  for (int v = 0; v < 8; ++v) {
    float* cr0 = Cb + (size_t)(v + mo) * ldc;
    float* cr1 = Cb + (size_t)(16 + v + mo) * ldc;
    cr0[n]      = c00[v];
    cr0[16 + n] = c01[v];
    cr0[32 + n] = c02[v];
    cr0[48 + n] = c03[v];
    cr1[n]      = c10[v];
    cr1[16 + n] = c11[v];
    cr1[32 + n] = c12[v];
    cr1[48 + n] = c13[v];
  }
}

// ------------------------- elementwise / prep kernels -----------------------

// dst (rows x Cp) bf16 = zero-padded copy of src (rows x C0) f32.
__global__ void k_pad_convert(const float* __restrict__ src, int C0, int Cp,
                              unsigned short* __restrict__ dst, long long total) {
  long long t = (long long)blockIdx.x * 256 + threadIdx.x;
  if (t >= total) return;
  long long r = t / Cp;
  int c = (int)(t % Cp);
  dst[t] = f2bf(c < C0 ? src[r * C0 + c] : 0.f);
}

// dst (Np x Kp) bf16 = transpose of src (K0 x N0) f32, zero-padded.
__global__ void k_transpose_convert(const float* __restrict__ src, int K0, int N0,
                                    int Kp, int Np, unsigned short* __restrict__ dst) {
  long long t = (long long)blockIdx.x * 256 + threadIdx.x;
  long long total = (long long)Kp * Np;
  if (t >= total) return;
  int n = (int)(t / Kp);
  int k = (int)(t % Kp);
  float v = (n < N0 && k < K0) ? src[(size_t)k * N0 + n] : 0.f;
  dst[t] = f2bf(v);
}

// W_heads (H,F,NHID) -> WcatT (512 x FPAD): WcatT[h*64+d][f] = W[h][f][d].
__global__ void k_wheads_T(const float* __restrict__ W, unsigned short* __restrict__ dst) {
  long long t = (long long)blockIdx.x * 256 + threadIdx.x;
  if (t >= (long long)HNH * FPAD) return;
  int n = (int)(t / FPAD);
  int k = (int)(t % FPAD);
  int h = n >> 6, d = n & 63;
  float v = (k < Fdim) ? W[((size_t)h * Fdim + k) * 64 + d] : 0.f;
  dst[t] = f2bf(v);
}

// s,d per (b,h,n): dot of Wh row slice with a_heads halves.
__global__ void k_sd(const float* __restrict__ Wh, const float* __restrict__ a_heads,
                     float* __restrict__ s, float* __restrict__ d) {
  int t = blockIdx.x * 256 + threadIdx.x;
  if (t >= Bsz * Hh * Natom) return;
  int n = t % Natom;
  int bh = t / Natom;
  int h = bh % Hh, b = bh / Hh;
  const float* w  = Wh + ((size_t)(b * Natom + n)) * HNH + h * 64;
  const float* a1 = a_heads + h * 128;
  float ss = 0.f, dd = 0.f;
#pragma unroll 8
  for (int k = 0; k < 64; ++k) { float x = w[k]; ss += x * a1[k]; dd += x * a1[64 + k]; }
  s[t] = ss; d[t] = dd;
}

// s2,d2 per (b,n) over HID cols of Wh2.
__global__ void k_sd2(const float* __restrict__ Wh2, const float* __restrict__ a_out,
                      float* __restrict__ s2, float* __restrict__ d2) {
  int t = blockIdx.x * 256 + threadIdx.x;
  if (t >= Bsz * Natom) return;
  const float* w = Wh2 + (size_t)t * HIDP;
  float ss = 0.f, dd = 0.f;
  for (int k = 0; k < HID; ++k) { float x = w[k]; ss += x * a_out[k]; dd += x * a_out[HID + k]; }
  s2[t] = ss; d2[t] = dd;
}

// Per-batch transpose to bf16: dst[b][c][j] = src[(b*Natom+j)*C + c].
__global__ void k_bT(const float* __restrict__ src, int C, unsigned short* __restrict__ dst) {
  long long t = (long long)blockIdx.x * 256 + threadIdx.x;
  long long total = (long long)Bsz * C * Natom;
  if (t >= total) return;
  int j = (int)(t % Natom);
  long long r = t / Natom;
  int c = (int)(r % C);
  int b = (int)(r / C);
  dst[t] = f2bf(src[((size_t)(b * Natom + j)) * C + c]);
}

// Masked leaky-relu attention logits + row softmax -> bf16 attention matrix.
// One wave32 per row of 128. heads = 8 (layer 1) or 1 (layer 2).
__global__ void k_attn(const float* __restrict__ s, const float* __restrict__ d,
                       const int* __restrict__ adj, unsigned short* __restrict__ attn,
                       int heads) {
  int w = blockIdx.x * (blockDim.x >> 5) + (threadIdx.x >> 5);
  if (w >= Bsz * heads * Natom) return;
  int lane = threadIdx.x & 31;
  int i  = w % Natom;
  int bh = w / Natom;
  int b  = bh / heads;
  float si = s[(size_t)bh * Natom + i];
  const float* dr = d + (size_t)bh * Natom;
  const int*   ar = adj + ((size_t)b * Natom + i) * Natom;
  float v[4];
  float mx = -3.4e38f;
#pragma unroll
  for (int t = 0; t < 4; ++t) {
    int j = lane + t * 32;
    float e = si + dr[j];
    e = e > 0.f ? e : 0.2f * e;          // LeakyReLU(alpha=0.2)
    e = (ar[j] > 0) ? e : NEGV;          // adjacency mask
    v[t] = e;
    mx = fmaxf(mx, e);
  }
#pragma unroll
  for (int off = 16; off > 0; off >>= 1) mx = fmaxf(mx, __shfl_xor(mx, off, 32));
  float sum = 0.f;
#pragma unroll
  for (int t = 0; t < 4; ++t) { v[t] = __expf(v[t] - mx); sum += v[t]; }
#pragma unroll
  for (int off = 16; off > 0; off >>= 1) sum += __shfl_xor(sum, off, 32);
  float inv = 1.f / sum;
  unsigned short* o = attn + (size_t)w * Natom;
#pragma unroll
  for (int t = 0; t < 4; ++t) o[lane + t * 32] = f2bf(v[t] * inv);
}

// elu + f32 -> bf16 (for hcat feeding GEMM3).
__global__ void k_elu_bf(const float* __restrict__ in, unsigned short* __restrict__ out,
                         long long total) {
  long long t = (long long)blockIdx.x * 256 + threadIdx.x;
  if (t >= total) return;
  float x = in[t];
  x = x > 0.f ? x : (__expf(x) - 1.f);
  out[t] = f2bf(x);
}

// logsumexp over elu(row[0:HID]) — one wave per row (row stride HIDP).
__global__ void k_lse(const float* __restrict__ out2, float* __restrict__ lse) {
  int w = blockIdx.x * (blockDim.x >> 5) + (threadIdx.x >> 5);
  if (w >= Bsz * Natom) return;
  int lane = threadIdx.x & 31;
  const float* row = out2 + (size_t)w * HIDP;
  float mx = -3.4e38f;
  for (int c = lane; c < HID; c += 32) {
    float x = row[c]; x = x > 0.f ? x : (__expf(x) - 1.f);
    mx = fmaxf(mx, x);
  }
#pragma unroll
  for (int off = 16; off > 0; off >>= 1) mx = fmaxf(mx, __shfl_xor(mx, off, 32));
  float sm = 0.f;
  for (int c = lane; c < HID; c += 32) {
    float x = row[c]; x = x > 0.f ? x : (__expf(x) - 1.f);
    sm += __expf(x - mx);
  }
#pragma unroll
  for (int off = 16; off > 0; off >>= 1) sm += __shfl_xor(sm, off, 32);
  if (lane == 0) lse[w] = mx + __logf(sm);
}

// mean_n( elu(out2[b,n,c]) - lse[b,n] ) -> gat_bf (256 x HIDP bf16, zero-padded).
__global__ void k_pool(const float* __restrict__ out2, const float* __restrict__ lse,
                       unsigned short* __restrict__ gat_bf) {
  int t = blockIdx.x * 256 + threadIdx.x;
  if (t >= Bsz * HIDP) return;
  int c = t % HIDP, b = t / HIDP;
  float acc = 0.f;
  if (c < HID) {
    const float* col = out2 + (size_t)b * Natom * HIDP + c;
    const float* l   = lse + (size_t)b * Natom;
    for (int n = 0; n < Natom; ++n) {
      float x = col[(size_t)n * HIDP];
      x = x > 0.f ? x : (__expf(x) - 1.f);
      acc += x - l[n];
    }
    acc *= (1.f / Natom);
  }
  gat_bf[t] = f2bf(acc);
}

// out[r*ldout + col0 + c] = bf16( act(in[r*ldin+c] + bias[c]) ) for c<nb else 0.
__global__ void k_biasact(const float* __restrict__ in, int ldin,
                          const float* __restrict__ bias, int nb, int act,
                          unsigned short* __restrict__ out, int ldout,
                          int col0, int ncols, int rows) {
  int t = blockIdx.x * 256 + threadIdx.x;
  if (t >= rows * ncols) return;
  int c = t % ncols, r = t / ncols;
  float v = 0.f;
  if (c < nb) {
    v = in[(size_t)r * ldin + c] + bias[c];
    if (act) v = fmaxf(v, 0.f);
  }
  out[(size_t)r * ldout + col0 + c] = f2bf(v);
}

// sigmoid( relu(y1 + b1) . w2 + b2 ) per batch row.
__global__ void k_final(const float* __restrict__ y1, const float* __restrict__ b1,
                        const float* __restrict__ w2, const float* __restrict__ b2,
                        float* __restrict__ out) {
  int b = blockIdx.x * blockDim.x + threadIdx.x;
  if (b >= Bsz) return;
  const float* r = y1 + (size_t)b * HIDP;
  float acc = b2[0];
  for (int c = 0; c < HID; ++c) acc += fmaxf(r[c] + b1[c], 0.f) * w2[c];
  out[b] = 1.f / (1.f + __expf(-acc));
}

// ---------------------------------------------------------------------------

#define CEILD(a, b) (((a) + (b) - 1) / (b))

extern "C" void kernel_launch(void* const* d_in, const int* in_sizes, int n_in,
                              void* d_out, int out_size, void* d_ws, size_t ws_size,
                              hipStream_t stream) {
  (void)in_sizes; (void)n_in; (void)out_size; (void)ws_size;

  const float* atom    = (const float*)d_in[0];
  const float* fp      = (const float*)d_in[1];
  const float* W_heads = (const float*)d_in[2];
  const float* a_heads = (const float*)d_in[3];
  const float* W_out   = (const float*)d_in[4];
  const float* a_out   = (const float*)d_in[5];
  const float* fc1_w   = (const float*)d_in[6];
  const float* fc1_b   = (const float*)d_in[7];
  const float* fc2_w   = (const float*)d_in[8];
  const float* fc2_b   = (const float*)d_in[9];
  const float* fcg_w   = (const float*)d_in[10];
  const float* fcg_b   = (const float*)d_in[11];
  const float* fcf_w   = (const float*)d_in[12];
  const float* fcf_b   = (const float*)d_in[13];
  const float* ffn_w1  = (const float*)d_in[14];
  const float* ffn_b1  = (const float*)d_in[15];
  const float* ffn_w2  = (const float*)d_in[16];
  const float* ffn_b2  = (const float*)d_in[17];
  const int*   adj     = (const int*)d_in[18];

  typedef unsigned short us;
  const long long BN = (long long)Bsz * Natom;   // 32768

  char* w = (char*)d_ws;
  size_t off = 0;
  auto alloc = [&](size_t bytes) -> void* {
    off = (off + 255) & ~(size_t)255;
    void* p = w + off;
    off += bytes;
    return p;
  };

  us* Xbf    = (us*)alloc(BN * FPAD * 2);            // 32768 x 160
  us* WcatT  = (us*)alloc((size_t)HNH * FPAD * 2);   // 512 x 160
  float* Wh  = (float*)alloc(BN * HNH * 4);          // 32768 x 512 (also hcat_f32)
  float* sb  = (float*)alloc((size_t)Bsz * Hh * Natom * 4);
  float* db  = (float*)alloc((size_t)Bsz * Hh * Natom * 4);
  us* WhT    = (us*)alloc((size_t)Bsz * HNH * Natom * 2);
  us* attn1b = (us*)alloc((size_t)Bsz * Hh * Natom * Natom * 2);
  us* hcatbf = (us*)alloc(BN * HNH * 2);
  us* WoutT  = (us*)alloc((size_t)HIDP * HNH * 2);
  float* Wh2 = (float*)alloc(BN * HIDP * 4);         // 32768 x 320 (also out2)
  float* s2  = (float*)alloc(BN * 4);
  float* d2  = (float*)alloc(BN * 4);
  us* Wh2T   = (us*)alloc((size_t)Bsz * HIDP * Natom * 2);
  us* attn2b = (us*)alloc((size_t)Bsz * Natom * Natom * 2);
  float* lse = (float*)alloc(BN * 4);
  us* gatbf  = (us*)alloc((size_t)Bsz * HIDP * 2);
  us* fpbf   = (us*)alloc((size_t)Bsz * FPPAD * 2);
  us* fc1T   = (us*)alloc((size_t)512 * FPPAD * 2);
  float* fpn1 = (float*)alloc((size_t)Bsz * 512 * 4);
  us* fpn1bf = (us*)alloc((size_t)Bsz * 512 * 2);
  us* fc2T   = (us*)alloc((size_t)HIDP * 512 * 2);
  float* fpn2 = (float*)alloc((size_t)Bsz * HIDP * 4);
  us* fpn2bf = (us*)alloc((size_t)Bsz * HIDP * 2);
  us* fcfT   = (us*)alloc((size_t)HIDP * HIDP * 2);
  float* fpn3 = (float*)alloc((size_t)Bsz * HIDP * 4);
  us* fcgT   = (us*)alloc((size_t)HIDP * HIDP * 2);
  float* gat2 = (float*)alloc((size_t)Bsz * HIDP * 4);
  us* xcat   = (us*)alloc((size_t)Bsz * CATP * 2);
  us* ffn1T  = (us*)alloc((size_t)HIDP * CATP * 2);
  float* y1  = (float*)alloc((size_t)Bsz * HIDP * 4);

  float* hcat = Wh;    // alias: Wh f32 dead after k_sd / k_bT
  float* out2 = Wh2;   // alias: Wh2 f32 dead after k_sd2 / k_bT

  // ---- GAT layer 1 ----
  k_pad_convert<<<CEILD(BN * FPAD, 256), 256, 0, stream>>>(atom, Fdim, FPAD, Xbf, BN * FPAD);
  k_wheads_T<<<CEILD((long long)HNH * FPAD, 256), 256, 0, stream>>>(W_heads, WcatT);
  // Wh = X @ Wcat : M=32768 N=512 K=160
  wmma_gemm<<<dim3(HNH / 64, (unsigned)(BN / 128), 1), 128, 0, stream>>>(
      Xbf, FPAD, 0, 0, WcatT, FPAD, 0, 0, Wh, HNH, 0, 0, FPAD, 1);
  k_sd<<<CEILD(Bsz * Hh * Natom, 256), 256, 0, stream>>>(Wh, a_heads, sb, db);
  k_bT<<<CEILD((long long)Bsz * HNH * Natom, 256), 256, 0, stream>>>(Wh, HNH, WhT);
  k_attn<<<CEILD(Bsz * Hh * Natom, 8), 256, 0, stream>>>(sb, db, adj, attn1b, Hh);
  // hcat[b, :, h*64:(h+1)*64] = attn[b,h] @ Wh[b,h] : batch 2048, M=128 N=64 K=128
  wmma_gemm<<<dim3(1, 1, Bsz * Hh), 128, 0, stream>>>(
      attn1b, Natom, (long long)Hh * Natom * Natom, (long long)Natom * Natom,
      WhT, Natom, (long long)HNH * Natom, 64LL * Natom,
      hcat, HNH, (long long)Natom * HNH, 64, Natom, Hh);
  k_elu_bf<<<CEILD(BN * HNH, 256), 256, 0, stream>>>(hcat, hcatbf, BN * HNH);

  // ---- GAT layer 2 ----
  k_transpose_convert<<<CEILD((long long)HIDP * HNH, 256), 256, 0, stream>>>(
      W_out, HNH, HID, HNH, HIDP, WoutT);
  // Wh2 = hcat @ W_out : M=32768 N=320 K=512
  wmma_gemm<<<dim3(HIDP / 64, (unsigned)(BN / 128), 1), 128, 0, stream>>>(
      hcatbf, HNH, 0, 0, WoutT, HNH, 0, 0, Wh2, HIDP, 0, 0, HNH, 1);
  k_sd2<<<CEILD((int)BN, 256), 256, 0, stream>>>(Wh2, a_out, s2, d2);
  k_bT<<<CEILD((long long)Bsz * HIDP * Natom, 256), 256, 0, stream>>>(Wh2, HIDP, Wh2T);
  k_attn<<<CEILD((int)BN, 8), 256, 0, stream>>>(s2, d2, adj, attn2b, 1);
  // out2[b] = attn2[b] @ Wh2[b] : batch 256, M=128 N=320 K=128
  wmma_gemm<<<dim3(HIDP / 64, 1, Bsz), 128, 0, stream>>>(
      attn2b, Natom, (long long)Natom * Natom, 0,
      Wh2T, Natom, (long long)HIDP * Natom, 0,
      out2, HIDP, (long long)Natom * HIDP, 0, Natom, 1);
  k_lse<<<CEILD((int)BN, 8), 256, 0, stream>>>(out2, lse);
  k_pool<<<CEILD(Bsz * HIDP, 256), 256, 0, stream>>>(out2, lse, gatbf);

  // ---- FPN branch ----
  k_pad_convert<<<CEILD((long long)Bsz * FPPAD, 256), 256, 0, stream>>>(
      fp, FP_DIM, FPPAD, fpbf, (long long)Bsz * FPPAD);
  k_transpose_convert<<<CEILD((long long)512 * FPPAD, 256), 256, 0, stream>>>(
      fc1_w, FP_DIM, 512, FPPAD, 512, fc1T);
  wmma_gemm<<<dim3(512 / 64, Bsz / 128, 1), 128, 0, stream>>>(
      fpbf, FPPAD, 0, 0, fc1T, FPPAD, 0, 0, fpn1, 512, 0, 0, FPPAD, 1);
  k_biasact<<<CEILD(Bsz * 512, 256), 256, 0, stream>>>(
      fpn1, 512, fc1_b, 512, 1, fpn1bf, 512, 0, 512, Bsz);
  k_transpose_convert<<<CEILD((long long)HIDP * 512, 256), 256, 0, stream>>>(
      fc2_w, 512, HID, 512, HIDP, fc2T);
  wmma_gemm<<<dim3(HIDP / 64, Bsz / 128, 1), 128, 0, stream>>>(
      fpn1bf, 512, 0, 0, fc2T, 512, 0, 0, fpn2, HIDP, 0, 0, 512, 1);
  k_biasact<<<CEILD(Bsz * HIDP, 256), 256, 0, stream>>>(
      fpn2, HIDP, fc2_b, HID, 0, fpn2bf, HIDP, 0, HIDP, Bsz);
  k_transpose_convert<<<CEILD((long long)HIDP * HIDP, 256), 256, 0, stream>>>(
      fcf_w, HID, HID, HIDP, HIDP, fcfT);
  wmma_gemm<<<dim3(HIDP / 64, Bsz / 128, 1), 128, 0, stream>>>(
      fpn2bf, HIDP, 0, 0, fcfT, HIDP, 0, 0, fpn3, HIDP, 0, 0, HIDP, 1);
  k_biasact<<<CEILD(Bsz * 308, 256), 256, 0, stream>>>(
      fpn3, HIDP, fcf_b, HID, 1, xcat, CATP, 300, 308, Bsz);  // cols 300..607

  // ---- GAT head FC + concat ----
  k_transpose_convert<<<CEILD((long long)HIDP * HIDP, 256), 256, 0, stream>>>(
      fcg_w, HID, HID, HIDP, HIDP, fcgT);
  wmma_gemm<<<dim3(HIDP / 64, Bsz / 128, 1), 128, 0, stream>>>(
      gatbf, HIDP, 0, 0, fcgT, HIDP, 0, 0, gat2, HIDP, 0, 0, HIDP, 1);
  k_biasact<<<CEILD(Bsz * 300, 256), 256, 0, stream>>>(
      gat2, HIDP, fcg_b, HID, 1, xcat, CATP, 0, 300, Bsz);    // cols 0..299

  // ---- Fusion FFN ----
  k_transpose_convert<<<CEILD((long long)HIDP * CATP, 256), 256, 0, stream>>>(
      ffn_w1, 600, HID, CATP, HIDP, ffn1T);
  wmma_gemm<<<dim3(HIDP / 64, Bsz / 128, 1), 128, 0, stream>>>(
      xcat, CATP, 0, 0, ffn1T, CATP, 0, 0, y1, HIDP, 0, 0, CATP, 1);
  k_final<<<1, 256, 0, stream>>>(y1, ffn_b1, ffn_w2, ffn_b2, (float*)d_out);
}
